// SafeGRU_46033459479129
// MI455X (gfx1250) — matrix operand order repaired and checked by reading, not compiled, not run on hardware
//
#include <hip/hip_runtime.h>

// GRU on MI455X (gfx1250, wave32, WMMA + async-LDS).
// Phase 0: fp32 -> bf16 conversion of x, Wx, Wh into workspace.
// Phase 1: x_lin = x @ Wx^T + bx (bf16 WMMA, f32 accum) -> bf16 in workspace.
// Phase 2: persistent scan, batch-partitioned (no inter-WG sync):
//          8 blocks x 16 batch rows; h in LDS (f32 + bf16 mirror); Wh from L2;
//          x_lin tiles double-buffered into LDS with GLOBAL_LOAD_ASYNC_TO_LDS_B128
//          overlapped with the recurrent WMMA GEMM; K loops fully unrolled so the
//          scheduler software-pipelines loads over WMMAs.

typedef __bf16 bf16_t;
typedef __attribute__((ext_vector_type(16))) __bf16 v16bf;
typedef __attribute__((ext_vector_type(8)))  __bf16 v8bf;
typedef __attribute__((ext_vector_type(8)))  float  v8f;

#define GRU_B  128
#define GRU_T  512
#define GRU_I  512
#define GRU_H  512
#define GRU_3H 1536

__device__ __forceinline__ bf16_t f32_to_bf16(float f) {
  union { float f; unsigned u; } cv; cv.f = f;
  unsigned u = cv.u;
  unsigned r = u + 0x7FFFu + ((u >> 16) & 1u);   // round-to-nearest-even
  union { unsigned short s; bf16_t b; } cb;
  cb.s = (unsigned short)(r >> 16);
  return cb.b;
}

// A-fragment (16x32 bf16, MxK): lane = M (l&15), K sub-base = (l>>4)*16.
// v16bf half 0 <- K = sub..sub+7, half 1 <- K = sub+8..sub+15.
// Union of halves: two b128 loads straight into adjacent VGPRs, no per-element movs.
__device__ __forceinline__ v16bf load_frag_a(const bf16_t* rowp, int sub) {
  union { v16bf v; v8bf h[2]; } u;
  u.h[0] = *(const v8bf*)(rowp + sub);
  u.h[1] = *(const v8bf*)(rowp + sub + 8);
  return u.v;
}

// CDNA5 async copy: 16 bytes global -> LDS per lane, tracked by ASYNCcnt.
__device__ __forceinline__ void async_g2l_b128(unsigned lds_off, const void* gptr) {
  asm volatile("global_load_async_to_lds_b128 %0, %1, off"
               :: "v"(lds_off), "v"((unsigned long long)(size_t)gptr)
               : "memory");
}
__device__ __forceinline__ void wait_async0() {
  asm volatile("s_wait_asynccnt 0x0" ::: "memory");
}

// ---------------------------------------------------------------- Phase 0
__global__ void __launch_bounds__(256)
cvt_f32_bf16(const float* __restrict__ src, bf16_t* __restrict__ dst, int n) {
  int i = blockIdx.x * 256 + threadIdx.x;
  if (i < n) dst[i] = f32_to_bf16(src[i]);
}

// ---------------------------------------------------------------- Phase 1
// Xb  : [B*T, I]  bf16 (row-major, K contiguous)
// Wxb : [3H, I]   bf16 (row-major, K contiguous)  -> "NT" GEMM
// xlin: [B*T, 3H] bf16
// Block = 128 threads (4 waves). Wave computes a 16x128 tile (8 WMMA accums).
__global__ void __launch_bounds__(128)
xlin_gemm(const bf16_t* __restrict__ Xb, const bf16_t* __restrict__ Wxb,
          const float* __restrict__ bx, bf16_t* __restrict__ xlin) {
  const int lane = threadIdx.x & 31;
  const int wave = threadIdx.x >> 5;
  const int m0 = blockIdx.x * 16;
  const int n0 = (blockIdx.y * 4 + wave) * 128;

  const int am  = lane & 15;          // A: lane -> M
  const int akb = (lane >> 4) * 16;   // A: K sub-base
  const int bn  = lane & 15;          // B: lane -> N
  const int bkb = (lane >> 4) * 16;   // B: K sub-base

  const bf16_t* arow = Xb + (size_t)(m0 + am) * GRU_I;
  __builtin_prefetch(arow + 128, 0, 3);   // cover the A row's later cachelines
  __builtin_prefetch(arow + 384, 0, 3);

  // Hoist per-tile B base pointers out of the K loop.
  const bf16_t* bbase[8];
#pragma unroll
  for (int t = 0; t < 8; ++t)
    bbase[t] = Wxb + (size_t)(n0 + t * 16 + bn) * GRU_I + bkb;

  v8f acc[8] = {};

#pragma unroll
  for (int ks = 0; ks < GRU_I / 32; ++ks) {   // fully unrolled: scheduler pipelines
    const int k = ks * 32;
    v16bf a = load_frag_a(arow + k, akb);
    v16bf bfrag[8];
#pragma unroll
    for (int t = 0; t < 8; ++t)
      bfrag[t] = *(const v16bf*)(bbase[t] + k);
#pragma unroll
    for (int t = 0; t < 8; ++t)
      acc[t] = __builtin_amdgcn_wmma_f32_16x16x32_bf16(
          false, a, false, bfrag[t], (short)0, acc[t], false, false);
  }

  // C/D layout: reg r, lane l -> M = r + 8*(l>>4), N = l&15
  const int mrow = (lane >> 4) * 8;
  const int nc   = lane & 15;
#pragma unroll
  for (int t = 0; t < 8; ++t) {
    const int col = n0 + t * 16 + nc;
    const float bias = bx[col];
#pragma unroll
    for (int r = 0; r < 8; ++r) {
      xlin[(size_t)(m0 + mrow + r) * GRU_3H + col] = f32_to_bf16(acc[t][r] + bias);
    }
  }
}

// ---------------------------------------------------------------- Phase 2
// Persistent scan: grid = 8 blocks (16 batch rows each), block = 512 thr (16 waves).
// Wave w owns h columns [w*32, w*32+32): 6 accum tiles = {z,r,n} x 2.
__global__ void __launch_bounds__(512)
gru_scan(const bf16_t* __restrict__ Whb,   // [3H, H] bf16, K contiguous
         const bf16_t* __restrict__ xlin,  // [B*T, 3H] bf16
         const float*  __restrict__ h0,    // [B, H] f32
         float* __restrict__ y,            // [B, T, H] f32
         float* __restrict__ hfin) {       // [B, H] f32
  __shared__ __align__(16) float  hF[16][GRU_H];        // 32 KB
  __shared__ __align__(16) bf16_t hB[16][GRU_H];        // 16 KB
  __shared__ __align__(16) bf16_t xs[2][16 * GRU_3H];   // 2 x 48 KB async stage

  const int tid  = threadIdx.x;
  const int lane = tid & 31;
  const int wave = tid >> 5;            // 0..15
  const int b0   = blockIdx.x * 16;     // batch rows [b0, b0+16)

  for (int i = tid; i < 16 * GRU_H; i += 512) {
    const int m = i >> 9, c = i & (GRU_H - 1);
    const float v = h0[(size_t)(b0 + m) * GRU_H + c];
    hF[m][c] = v;
    hB[m][c] = f32_to_bf16(v);
  }

  // Per-thread async-stage chunks: 16*1536 bf16 = 3072 x b128; 6 chunks/thread.
  // chunk c -> row m = c/192, 16-elem group g = c%192.
  unsigned xs_off[2];
  xs_off[0] = (unsigned)(size_t)&xs[0][0];
  xs_off[1] = (unsigned)(size_t)&xs[1][0];

  // Prologue: stage x_lin tile for t=0 into buffer 0.
#pragma unroll
  for (int i = 0; i < 6; ++i) {
    const int c = tid + i * 512;
    const int m = c / 192, g = c % 192;
    const bf16_t* src = xlin + ((size_t)(b0 + m) * GRU_T + 0) * GRU_3H + g * 16;
    async_g2l_b128(xs_off[0] + (unsigned)c * 32u, src);
  }
  wait_async0();
  __syncthreads();

  const int am    = lane & 15;
  const int akb   = (lane >> 4) * 16;
  const int bn    = lane & 15;
  const int bkb   = (lane >> 4) * 16;
  const int mrow  = (lane >> 4) * 8;
  const int nc    = lane & 15;
  const int ncol0 = wave * 32;          // this wave's output columns

  // Hoist the 6 Wh tile base pointers (z0,z1,r0,r1,n0,n1).
  const bf16_t* whbase[6];
#pragma unroll
  for (int j = 0; j < 6; ++j) {
    const int g = j >> 1, tt = j & 1;
    whbase[j] = Whb + (size_t)(g * GRU_H + ncol0 + tt * 16 + bn) * GRU_H + bkb;
  }

  for (int t = 0; t < GRU_T; ++t) {
    const int buf = t & 1;

    // Kick off async staging of x_lin for step t+1 (overlaps the GEMM below).
    if (t + 1 < GRU_T) {
#pragma unroll
      for (int i = 0; i < 6; ++i) {
        const int c = tid + i * 512;
        const int m = c / 192, g = c % 192;
        const bf16_t* src =
            xlin + ((size_t)(b0 + m) * GRU_T + (t + 1)) * GRU_3H + g * 16;
        async_g2l_b128(xs_off[buf ^ 1] + (unsigned)c * 32u, src);
      }
    }

    v8f acc[6] = {};                    // [z0,z1,r0,r1,n0,n1]
#pragma unroll
    for (int ks = 0; ks < GRU_H / 32; ++ks) {   // fully unrolled critical path
      const int k = ks * 32;
      v16bf a = load_frag_a(&hB[am][k], akb);   // A = h (LDS, bf16)
      v16bf bfrag[6];
#pragma unroll
      for (int j = 0; j < 6; ++j)
        bfrag[j] = *(const v16bf*)(whbase[j] + k);
#pragma unroll
      for (int j = 0; j < 6; ++j)
        acc[j] = __builtin_amdgcn_wmma_f32_16x16x32_bf16(
            false, a, false, bfrag[j], (short)0, acc[j], false, false);
    }

    // Gates: x_lin comes from the LDS stage filled during step t-1.
    const bf16_t* xrow = &xs[buf][0];
    float hnew_s[2][8];
#pragma unroll
    for (int tt = 0; tt < 2; ++tt) {
      const int col = ncol0 + tt * 16 + nc;
#pragma unroll
      for (int r = 0; r < 8; ++r) {
        const int m = mrow + r;
        const int xb = m * GRU_3H + col;
        const float xz = (float)xrow[xb];
        const float xr = (float)xrow[xb + GRU_H];
        const float xn = (float)xrow[xb + 2 * GRU_H];
        const float z  = 1.0f / (1.0f + __expf(-(xz + acc[tt][r])));
        const float rr = 1.0f / (1.0f + __expf(-(xr + acc[2 + tt][r])));
        const float nn = tanhf(xn + rr * acc[4 + tt][r]);
        const float hp = hF[m][col];
        const float hnew = (1.0f - z) * nn + z * hp;
        hnew_s[tt][r] = hnew;
        y[((size_t)(b0 + m) * GRU_T + t) * GRU_H + col] = hnew;
      }
    }

    wait_async0();                      // t+1 stage landed in LDS
    __syncthreads();                    // all waves done reading hB/hF/xs
#pragma unroll
    for (int tt = 0; tt < 2; ++tt) {
      const int col = ncol0 + tt * 16 + nc;
#pragma unroll
      for (int r = 0; r < 8; ++r) {
        const int m = mrow + r;
        hF[m][col] = hnew_s[tt][r];
        hB[m][col] = f32_to_bf16(hnew_s[tt][r]);
      }
    }
    __syncthreads();                    // h updated for next step
  }

#pragma unroll
  for (int tt = 0; tt < 2; ++tt) {
    const int col = ncol0 + tt * 16 + nc;
#pragma unroll
    for (int r = 0; r < 8; ++r) {
      const int m = mrow + r;
      hfin[(size_t)(b0 + m) * GRU_H + col] = hF[m][col];
    }
  }
}

// ---------------------------------------------------------------- Launch
extern "C" void kernel_launch(void* const* d_in, const int* in_sizes, int n_in,
                              void* d_out, int out_size, void* d_ws, size_t ws_size,
                              hipStream_t stream) {
  (void)in_sizes; (void)n_in; (void)out_size; (void)ws_size;

  const float* x  = (const float*)d_in[0];   // [B,T,I]
  const float* h0 = (const float*)d_in[1];   // [B,H]
  const float* Wx = (const float*)d_in[2];   // [3H,I]
  const float* bx = (const float*)d_in[3];   // [3H]
  const float* Wh = (const float*)d_in[4];   // [3H,H]

  float* y    = (float*)d_out;                               // [B,T,H]
  float* hfin = y + (size_t)GRU_B * GRU_T * GRU_H;           // [B,H]

  // Workspace layout (~272 MB):
  //   xlin bf16 [B*T,3H] | x bf16 | Wx bf16 | Wh bf16
  char* ws = (char*)d_ws;
  bf16_t* xlin = (bf16_t*)ws;
  size_t off = (size_t)GRU_B * GRU_T * GRU_3H * sizeof(bf16_t);
  bf16_t* xb  = (bf16_t*)(ws + off); off += (size_t)GRU_B * GRU_T * GRU_I * sizeof(bf16_t);
  bf16_t* wxb = (bf16_t*)(ws + off); off += (size_t)GRU_3H * GRU_I * sizeof(bf16_t);
  bf16_t* whb = (bf16_t*)(ws + off);

  const int NX = GRU_B * GRU_T * GRU_I;   // 33,554,432
  const int NW = GRU_3H * GRU_I;          // 786,432

  cvt_f32_bf16<<<(NX + 255) / 256, 256, 0, stream>>>(x,  xb,  NX);
  cvt_f32_bf16<<<(NW + 255) / 256, 256, 0, stream>>>(Wx, wxb, NW);
  cvt_f32_bf16<<<(NW + 255) / 256, 256, 0, stream>>>(Wh, whb, NW);

  xlin_gemm<<<dim3((GRU_B * GRU_T) / 16, GRU_3H / 512, 1), 128, 0, stream>>>(
      xb, wxb, bx, xlin);

  gru_scan<<<dim3(GRU_B / 16, 1, 1), 512, 0, stream>>>(
      whb, xlin, h0, y, hfin);
}
